// DrugGNN_300647710826
// MI455X (gfx1250) — matrix operand order repaired and checked by reading, compiled, not run on hardware
//
#include <hip/hip_runtime.h>
#include <hip/hip_bf16.h>

#define N_NODES  131072
#define N_EDGES  524288
#define N_GRAPHS 4096
#define D_INF    64
#define D_HF     128

typedef __attribute__((ext_vector_type(16))) __bf16 v16bf;
typedef __attribute__((ext_vector_type(8)))  float  v8f;
typedef int v4i __attribute__((vector_size(16)));    // matches builtin param type

#define GLOBAL_AS __attribute__((address_space(1)))
#define LDS_AS    __attribute__((address_space(3)))

typedef GLOBAL_AS v4i* gv4ip;
typedef LDS_AS    v4i* lv4ip;

#if defined(__has_builtin)
#if __has_builtin(__builtin_amdgcn_global_load_async_to_lds_b128)
#define HAVE_ASYNC_LDS 1
#endif
#endif
#ifndef HAVE_ASYNC_LDS
#define HAVE_ASYNC_LDS 0
#endif

// ---------------- degree / normalization ----------------
__global__ void k_deg(const int* __restrict__ dst, float* __restrict__ deg) {
    int e = blockIdx.x * blockDim.x + threadIdx.x;
    if (e < N_EDGES) atomicAdd(&deg[dst[e]], 1.0f);
}

__global__ void k_dis(const float* __restrict__ deg, float* __restrict__ dis) {
    int i = blockIdx.x * blockDim.x + threadIdx.x;
    if (i < N_NODES) dis[i] = rsqrtf(deg[i] + 1.0f);
}

__global__ void k_coef(const int* __restrict__ src, const int* __restrict__ dst,
                       const float* __restrict__ dis, float* __restrict__ coef) {
    int e = blockIdx.x * blockDim.x + threadIdx.x;
    if (e < N_EDGES) coef[e] = dis[src[e]] * dis[dst[e]];
}

// ---------------- fused WMMA GEMM + self-loop/bias epilogue ----------------
// T = act(A) @ W ; agg = T*dis^2 + bias     (A: [N x K], W: [K x 128])
// Block: 256 threads = 8 waves. Block tile: 64 rows x 128 cols.
// Wave w owns cols [16w,16w+16) and 4 M-subtiles -> B frag reused 4x per K-step.
template<int K, bool RELU>
__global__ __launch_bounds__(256)
void k_gemm_wmma(const float* __restrict__ A, const float* __restrict__ Wm,
                 const float* __restrict__ bias, const float* __restrict__ dis,
                 float* __restrict__ T, float* __restrict__ agg) {
    constexpr int LDA = K + 4;                 // pad: (K+4)%64 != 0 -> conflict-free
    __shared__ float sA[64 * LDA];
    const int tid     = threadIdx.x;
    const int rowBase = blockIdx.x * 64;

    // ---- stage A tile (64 x K fp32) into LDS: pure b128 copy -> async path ----
    constexpr int V = K / 4;                   // float4 chunks per row
    for (int v = tid; v < 64 * V; v += 256) {
        int r = v / V, c4 = v - r * V;
        const float* gp = A + (size_t)(rowBase + r) * K + c4 * 4;
        float*       lp = &sA[r * LDA + c4 * 4];
#if HAVE_ASYNC_LDS
        __builtin_amdgcn_global_load_async_to_lds_b128((gv4ip)gp, (lv4ip)lp, 0, 0);
#else
        *(float4*)lp = *(const float4*)gp;
#endif
    }
#if HAVE_ASYNC_LDS
    asm volatile("s_wait_asynccnt 0" ::: "memory");
#endif
    __syncthreads();

    const int lane    = tid & 31;
    const int wave    = tid >> 5;
    const int m       = lane & 15;
    const int halfSel = lane >> 4;
    const int n       = wave * 16 + (lane & 15);

    v8f acc[4] = {};
#pragma unroll
    for (int kb = 0; kb < K; kb += 32) {
        // B 32x16 bf16 frag (from global; W is L2-hot): VGPR j -> K=2j+s, +16 for hi lanes
        v16bf b;
#pragma unroll
        for (int i = 0; i < 16; ++i) {
            int j = i >> 1, s = i & 1;
            int kk = kb + 2 * j + s + halfSel * 16;
            b[i] = (__bf16)Wm[(size_t)kk * D_HF + n];
        }
#pragma unroll
        for (int mt = 0; mt < 4; ++mt) {
            // A 16x32 bf16 frag: VGPR j -> K=2j+s, +16 for hi lanes
            v16bf a;
#pragma unroll
            for (int i = 0; i < 16; ++i) {
                int j = i >> 1, s = i & 1;
                int ka = kb + 2 * j + s + halfSel * 16;
                float v = sA[(mt * 16 + m) * LDA + ka];
                if (RELU) v = fmaxf(v, 0.0f);
                a[i] = (__bf16)v;
            }
            acc[mt] = __builtin_amdgcn_wmma_f32_16x16x32_bf16(
                false, a, false, b, (short)0, acc[mt], false, false);
        }
    }

    // ---- epilogue: T[o] = t ; agg[o] = t*dis^2 + bias (init for edge scatter) ----
    const float bn = bias[n];
#pragma unroll
    for (int mt = 0; mt < 4; ++mt) {
#pragma unroll
        for (int r = 0; r < 8; ++r) {
            int row  = rowBase + mt * 16 + r + 8 * halfSel;   // C/D layout
            float t  = acc[mt][r];
            float d  = dis[row];
            size_t o = (size_t)row * D_HF + n;
            T[o]   = t;
            agg[o] = t * d * d + bn;
        }
    }
}

// ---------------- edge scatter: agg[dst] += T[src] * coef ----------------
__global__ void k_edge_agg(const int* __restrict__ src, const int* __restrict__ dst,
                           const float* __restrict__ coef, const float* __restrict__ T,
                           float* __restrict__ agg) {
    int e    = blockIdx.x * 8 + (threadIdx.x >> 5);  // one wave per edge
    int lane = threadIdx.x & 31;
    if (e < N_EDGES) {
        int s = src[e], d = dst[e];
        float c = coef[e];
        const float* ts = T   + (size_t)s * D_HF;
        float*       ad = agg + (size_t)d * D_HF;
        __builtin_prefetch(ad + lane * 4, 1, 3);     // global_prefetch on atomic target
#pragma unroll
        for (int q = 0; q < 4; ++q) {
            int j = lane + 32 * q;
            atomicAdd(&ad[j], ts[j] * c);
        }
    }
}

// ---------------- global mean pool (ReLU of layer-3 fused here) ----------------
__global__ void k_pool(const float* __restrict__ h, const int* __restrict__ batch,
                       float* __restrict__ sums, float* __restrict__ cnt) {
    int i    = blockIdx.x * 8 + (threadIdx.x >> 5);  // one wave per node
    int lane = threadIdx.x & 31;
    if (i < N_NODES) {
        int g = batch[i];
#pragma unroll
        for (int q = 0; q < 4; ++q) {
            int j = lane + 32 * q;
            atomicAdd(&sums[(size_t)g * D_HF + j], fmaxf(h[(size_t)i * D_HF + j], 0.0f));
        }
        if (lane == 0) atomicAdd(&cnt[g], 1.0f);
    }
}

__global__ void k_final(const float* __restrict__ sums, const float* __restrict__ cnt,
                        float* __restrict__ out) {
    int t = blockIdx.x * blockDim.x + threadIdx.x;
    if (t < N_GRAPHS * D_HF) {
        int g = t >> 7;
        out[t] = sums[t] / fmaxf(cnt[g], 1.0f);
    }
}

// ---------------- driver ----------------
extern "C" void kernel_launch(void* const* d_in, const int* in_sizes, int n_in,
                              void* d_out, int out_size, void* d_ws, size_t ws_size,
                              hipStream_t stream) {
    const float* x     = (const float*)d_in[0];
    const int*   ei    = (const int*)d_in[1];
    const int*   batch = (const int*)d_in[2];
    const float* W1    = (const float*)d_in[3];
    const float* b1    = (const float*)d_in[4];
    const float* W2    = (const float*)d_in[5];
    const float* b2    = (const float*)d_in[6];
    const float* W3    = (const float*)d_in[7];
    const float* b3    = (const float*)d_in[8];
    float* out = (float*)d_out;

    const int* src = ei;
    const int* dst = ei + N_EDGES;

    float* ws = (float*)d_ws;
    size_t off = 0;
    float* T    = ws + off; off += (size_t)N_NODES * D_HF;
    float* H1   = ws + off; off += (size_t)N_NODES * D_HF;
    float* H2   = ws + off; off += (size_t)N_NODES * D_HF;
    float* dis  = ws + off; off += N_NODES;
    float* deg  = ws + off; off += N_NODES;
    float* coef = ws + off; off += N_EDGES;
    float* sums = ws + off; off += (size_t)N_GRAPHS * D_HF;
    float* cnt  = ws + off; off += N_GRAPHS;

    // accumulated-into buffers must be zeroed every call (graph-capture safe)
    (void)hipMemsetAsync(deg,  0, (size_t)N_NODES * sizeof(float), stream);
    (void)hipMemsetAsync(sums, 0, (size_t)N_GRAPHS * D_HF * sizeof(float), stream);
    (void)hipMemsetAsync(cnt,  0, (size_t)N_GRAPHS * sizeof(float), stream);

    k_deg <<<N_EDGES / 256, 256, 0, stream>>>(dst, deg);
    k_dis <<<N_NODES / 256, 256, 0, stream>>>(deg, dis);
    k_coef<<<N_EDGES / 256, 256, 0, stream>>>(src, dst, dis, coef);

    const int gGemm = N_NODES / 64;
    const int gEdge = N_EDGES / 8;

    // layer 1: x [N x 64] @ W1 -> T ; agg init fused -> H1 ; edge scatter
    k_gemm_wmma<D_INF, false><<<gGemm, 256, 0, stream>>>(x,  W1, b1, dis, T, H1);
    k_edge_agg<<<gEdge, 256, 0, stream>>>(src, dst, coef, T, H1);

    // layer 2: relu(H1) @ W2 -> T ; agg init -> H2 ; edge scatter
    k_gemm_wmma<D_HF, true><<<gGemm, 256, 0, stream>>>(H1, W2, b2, dis, T, H2);
    k_edge_agg<<<gEdge, 256, 0, stream>>>(src, dst, coef, T, H2);

    // layer 3: relu(H2) @ W3 -> T ; agg init -> H1 (reuse) ; edge scatter
    k_gemm_wmma<D_HF, true><<<gGemm, 256, 0, stream>>>(H2, W3, b3, dis, T, H1);
    k_edge_agg<<<gEdge, 256, 0, stream>>>(src, dst, coef, T, H1);

    // pooling (ReLU fused into accumulation)
    k_pool <<<N_NODES / 8, 256, 0, stream>>>(H1, batch, sums, cnt);
    k_final<<<(N_GRAPHS * D_HF) / 256, 256, 0, stream>>>(sums, cnt, out);
}
